// HighLowPassNet_84619445666494
// MI455X (gfx1250) — matrix operand order repaired and checked by reading, not compiled, not run on hardware
//
#include <hip/hip_runtime.h>
#include <math.h>

typedef __attribute__((ext_vector_type(16))) _Float16 v16h;
typedef __attribute__((ext_vector_type(8)))  _Float16 v8h;
typedef __attribute__((ext_vector_type(8)))  float    v8f;

#define TAPS 36

// ============================ utility kernels ============================
__global__ void k_fill_zero(float* __restrict__ p, long n) {
  long i = (long)blockIdx.x * blockDim.x + threadIdx.x;
  if (i < n) p[i] = 0.f;
}

__global__ void k_pack_y(const float* __restrict__ yri, float2* __restrict__ y, long n) {
  long i = (long)blockIdx.x * blockDim.x + threadIdx.x;
  if (i < n) y[i] = make_float2(yri[2 * i], yri[2 * i + 1]);
}

__global__ void k_copy_c(const float2* __restrict__ s, float2* __restrict__ d, long n) {
  long i = (long)blockIdx.x * blockDim.x + threadIdx.x;
  if (i < n) d[i] = s[i];
}

__global__ void k_gather_low(const float2* __restrict__ y, const int* __restrict__ lidx,
                             float2* __restrict__ ylow, int Ml, int M, int B) {
  long i = (long)blockIdx.x * blockDim.x + threadIdx.x;
  if (i >= (long)B * Ml) return;
  int b = (int)(i / Ml), m = (int)(i % Ml);
  ylow[i] = y[(long)b * M + lidx[m]];
}

__global__ void k_scatter_full(float2* __restrict__ full, const float2* __restrict__ mv,
                               const int* __restrict__ lidx, int Ml, int M, int B) {
  long i = (long)blockIdx.x * blockDim.x + threadIdx.x;
  if (i >= (long)B * Ml) return;
  int b = (int)(i / Ml), m = (int)(i % Ml);
  full[(long)b * M + lidx[m]] = mv[i];
}

__global__ void k_add(const float* __restrict__ a, const float* __restrict__ b,
                      float* __restrict__ o, long n) {
  long i = (long)blockIdx.x * blockDim.x + threadIdx.x;
  if (i < n) o[i] = a[i] + b[i];
}

// ==================== NUFFT gridding (adjoint scatter) ====================
__global__ void k_adj_scatter(const float2* __restrict__ y, const float* __restrict__ wgt,
                              const int* __restrict__ idx, const float* __restrict__ mw,
                              float2* __restrict__ grid, int M, int B, long KK) {
  long i = (long)blockIdx.x * blockDim.x + threadIdx.x;
  if (i >= (long)B * M) return;
  int b = (int)(i / M), m = (int)(i % M);
  float2 v = y[i];
  if (mw) { float wm = mw[m]; v.x *= wm; v.y *= wm; }
  float* g = (float*)(grid + (long)b * KK);
  #pragma unroll 4
  for (int t = 0; t < TAPS; ++t) {
    float w = wgt[(long)m * TAPS + t];
    long j = idx[(long)m * TAPS + t];
    atomicAdd(g + 2 * j, w * v.x);
    atomicAdd(g + 2 * j + 1, w * v.y);
  }
}

// ==================== NUFFT interpolation (direct gather) =================
__global__ void k_dir_gather(const float2* __restrict__ Xk, const float* __restrict__ wgt,
                             const int* __restrict__ idx, const float2* __restrict__ ysub,
                             float2* __restrict__ out, int M, int B, long KK) {
  long i = (long)blockIdx.x * blockDim.x + threadIdx.x;
  if (i >= (long)B * M) return;
  int b = (int)(i / M), m = (int)(i % M);
  const float2* X = Xk + (long)b * KK;
  __builtin_prefetch(&wgt[(long)m * TAPS], 0, 0);   // global_prefetch_b8
  float2 acc = make_float2(0.f, 0.f);
  #pragma unroll 4
  for (int t = 0; t < TAPS; ++t) {
    float w = wgt[(long)m * TAPS + t];
    int j = idx[(long)m * TAPS + t];
    float2 xv = X[j];
    acc.x += w * xv.x;
    acc.y += w * xv.y;
  }
  if (ysub) { float2 s = ysub[i]; acc.x -= s.x; acc.y -= s.y; }
  out[i] = acc;
}

// ============================ batched line FFT ============================
// Stockham radix-2 in LDS; one line per workgroup. sign=-1 fwd, +1 inv.
// Line staging uses CDNA5 async global->LDS DMA (ASYNCcnt tracked): each lane
// supplies its own global address and LDS byte offset; data bypasses VGPRs.
template <int L, int LOG2>
__global__ __launch_bounds__(L / 2) void k_fft_line(float2* __restrict__ data,
                                                    int lines_per_batch, long batch_stride,
                                                    long line_stride, long elem_stride,
                                                    float sgn, float scale) {
  __shared__ float2 buf[2][L];
  const int line = blockIdx.x;
  const long base = (long)(line / lines_per_batch) * batch_stride +
                    (long)(line % lines_per_batch) * line_stride;
  const int t = threadIdx.x;  // 0 .. L/2-1
  {
    unsigned l0 = (unsigned)(unsigned long long)&buf[0][t];
    unsigned l1 = (unsigned)(unsigned long long)&buf[0][t + L / 2];
    unsigned long long g0 = (unsigned long long)(data + base + (long)t * elem_stride);
    unsigned long long g1 = (unsigned long long)(data + base + (long)(t + L / 2) * elem_stride);
    asm volatile("global_load_async_to_lds_b64 %0, %1, off" :: "v"(l0), "v"(g0) : "memory");
    asm volatile("global_load_async_to_lds_b64 %0, %1, off" :: "v"(l1), "v"(g1) : "memory");
    asm volatile("s_wait_asynccnt 0x0" ::: "memory");
  }
  int src = 0, p = 1;
  for (int s = 0; s < LOG2; ++s, p <<= 1) {
    __syncthreads();
    const int k = t & (p - 1);
    const int g = t >> s;
    float2 u = buf[src][t];
    float2 v = buf[src][t + L / 2];
    float sw, cw;
    __sincosf(sgn * 3.14159265358979323846f * (float)k / (float)p, &sw, &cw);
    float2 vw = make_float2(v.x * cw - v.y * sw, v.x * sw + v.y * cw);
    const int dst = src ^ 1;
    const int o = (g << (s + 1)) + k;
    buf[dst][o] = make_float2(u.x + vw.x, u.y + vw.y);
    buf[dst][o + p] = make_float2(u.x - vw.x, u.y - vw.y);
    src = dst;
  }
  __syncthreads();
  float2 r0 = buf[src][t], r1 = buf[src][t + L / 2];
  data[base + (long)t * elem_stride] = make_float2(r0.x * scale, r0.y * scale);
  data[base + (long)(t + L / 2) * elem_stride] = make_float2(r1.x * scale, r1.y * scale);
}

// ===================== scaling / pad / crop helpers =======================
__global__ void k_pad_sn(const float* __restrict__ x, const float* __restrict__ sn,
                         float2* __restrict__ grid, int B, int N0, int N1, int K1, long KK,
                         int xstride, long xoff) {
  long i = (long)blockIdx.x * blockDim.x + threadIdx.x;
  if (i >= (long)B * N0 * N1) return;
  int b = (int)(i / ((long)N0 * N1));
  int r = (int)(i % ((long)N0 * N1));
  int i0 = r / N1, i1 = r % N1;
  float v = x[((long)b * N0 * N1 + r) * xstride + xoff] * sn[r];
  grid[(long)b * KK + (long)i0 * K1 + i1] = make_float2(v, 0.f);
}

__global__ void k_crop_sn_real(const float2* __restrict__ grid, const float* __restrict__ sn,
                               float* __restrict__ out, int B, int N0, int N1, int K1, long KK,
                               int ostride, long ooff) {
  long i = (long)blockIdx.x * blockDim.x + threadIdx.x;
  if (i >= (long)B * N0 * N1) return;
  int b = (int)(i / ((long)N0 * N1));
  int r = (int)(i % ((long)N0 * N1));
  int i0 = r / N1, i1 = r % N1;
  out[((long)b * N0 * N1 + r) * ostride + ooff] =
      grid[(long)b * KK + (long)i0 * K1 + i1].x * sn[r];
}

// =========================== BN / activations =============================
__global__ void k_bn_act(float* __restrict__ x, const float* __restrict__ g,
                         const float* __restrict__ bt, const float* __restrict__ mean,
                         const float* __restrict__ var, int C, long n, int relu_first) {
  long i = (long)blockIdx.x * blockDim.x + threadIdx.x;
  if (i >= n) return;
  int c = (int)(i % C);
  float v = x[i];
  if (relu_first) v = fmaxf(v, 0.f);
  x[i] = (v - mean[c]) * rsqrtf(var[c] + 1e-3f) * g[c] + bt[c];
}

__global__ void k_conv1x1_sig(const float* __restrict__ x, const float* __restrict__ k3,
                              const float* __restrict__ b3, float* __restrict__ out, int C,
                              long npix) {
  long i = (long)blockIdx.x * blockDim.x + threadIdx.x;
  if (i >= npix) return;
  float acc = b3[0];
  for (int c = 0; c < C; ++c) acc += x[i * C + c] * k3[c];
  out[i] = 1.f / (1.f + expf(-acc));
}

// HWIO f32 weights -> f16 in WMMA *fragment order*: one 32-byte contiguous run
// per (k0-block, outchannel, lane-half), so each lane's B fragment is a single
// v16h (2x global_load_b128) instead of 16 scalar u16 loads.
// d[(((k0/32)*COUT + co)*2 + half)*16 + kk] = w[(k0 + half*16 + kk)*COUT + co]
__global__ void k_cvt_w(const float* __restrict__ s, _Float16* __restrict__ d, int K, int KP,
                        int COUT) {
  long i = (long)blockIdx.x * blockDim.x + threadIdx.x;
  if (i >= (long)KP * COUT) return;
  int kk = (int)(i & 15);
  long r = i >> 4;
  int half = (int)(r & 1);
  r >>= 1;
  int co = (int)(r % COUT);
  int kb = (int)(r / COUT);
  int k = kb * 32 + half * 16 + kk;
  d[i] = (_Float16)(k < K ? s[(long)k * COUT + co] : 0.f);
}

// ================= 3x3 conv as implicit GEMM on WMMA ======================
// One wave computes a 16-pixel x 16-outchannel tile: D = A(16xKP) * B(KPx16).
// A im2col staged in LDS; per the 16-bit A layout each lane's fragment is two
// contiguous 16B runs -> 2x ds_load_b128. B is pre-swizzled -> 1x v16h load.
template <int CIN, int COUT, int KP>
__global__ __launch_bounds__(128) void k_conv3x3_wmma(const float* __restrict__ x,
                                                      const _Float16* __restrict__ wk,
                                                      const float* __restrict__ bias,
                                                      float* __restrict__ y, int B, int H,
                                                      int W) {
  constexpr int NT = COUT / 16;
  const int wave = threadIdx.x >> 5;
  const int lane = threadIdx.x & 31;
  const int mrow = lane & 15;     // M (pixel) for frag loads / N for C/D
  const int half = lane >> 4;     // lane-group select per ISA layout
  const int strips = H * (W / 16);
  const long gw = (long)blockIdx.x * 4 + wave;
  const long total = (long)B * strips * NT;
  __shared__ __align__(16) _Float16 lA[4][16 * KP];
  if (gw >= total) return;  // uniform per wave -> EXEC stays all-ones for WMMA

  const int nt = (int)(gw % NT);
  const long strip = gw / NT;
  const int b = (int)(strip / strips);
  const int rem = (int)(strip % strips);
  const int row = rem / (W / 16);
  const int col0 = (rem % (W / 16)) * 16;

  _Float16* A = lA[wave];
  // im2col: 16 pixels x KP (zero-padded K and SAME borders)
  for (int e = lane; e < 16 * KP; e += 32) {
    int m = e / KP, k = e % KP;
    float v = 0.f;
    if (k < 9 * CIN) {
      int dy = k / (3 * CIN), rk = k % (3 * CIN), dx = rk / CIN, ci = rk % CIN;
      int yy = row + dy - 1, xx = col0 + m + dx - 1;
      if (yy >= 0 && yy < H && xx >= 0 && xx < W)
        v = x[(((long)b * H + yy) * W + xx) * CIN + ci];
    }
    A[e] = (_Float16)v;
  }
  __builtin_amdgcn_wave_barrier();  // keep LDS fill ahead of fragment reads

  const v16h* Wf = (const v16h*)wk;
  v8f c = {};
  for (int k0 = 0; k0 < KP; k0 += 32) {
    // A fragment: K = k0 + half*16 + [0..15]
    const _Float16* Ap = A + mrow * KP + k0 + half * 16;
    v8h alo = *(const v8h*)Ap;
    v8h ahi = *(const v8h*)(Ap + 8);
    v16h a = __builtin_shufflevector(alo, ahi, 0, 1, 2, 3, 4, 5, 6, 7, 8, 9, 10, 11, 12, 13, 14,
                                     15);
    // B fragment: pre-swizzled contiguous 32 bytes per lane
    v16h bf = Wf[((long)(k0 >> 5) * COUT + nt * 16 + mrow) * 2 + half];
    c = __builtin_amdgcn_wmma_f32_16x16x32_f16(false, a, false, bf, (short)0, c, false, false);
  }
  __builtin_amdgcn_s_wait_tensorcnt(0);  // CDNA5 split-counter path (no-op here)

  // C/D layout: VGPR r -> M = r + 8*half ; N = lane&15
  #pragma unroll
  for (int r = 0; r < 8; ++r) {
    int m = r + 8 * half;
    int xx = col0 + m;
    y[(((long)b * H + row) * W + xx) * COUT + nt * 16 + mrow] = c[r] + bias[nt * 16 + mrow];
  }
}

// ============================== host side =================================
static inline dim3 gb(long n) { return dim3((unsigned)((n + 255) / 256)); }

extern "C" void kernel_launch(void* const* d_in, const int* in_sizes, int n_in, void* d_out,
                              int out_size, void* d_ws, size_t ws_size, hipStream_t stream) {
  (void)ws_size;
  const int B = 20, M = 50000;
  const long KK0 = 512L * 512L, KK1 = 256L * 256L;
  const long P0 = (long)B * 256 * 256;   // plan0 image pixels
  const long P1 = (long)B * 128 * 128;   // plan1 image pixels

  // -------- input unpack (leaf order of setup_inputs dict, depth-first) ----
  auto IN = [&](int i) -> const void* { return d_in[(i < n_in) ? i : 0]; };
  const float* y_ri = (const float*)IN(0);
  const int* idx0 = (const int*)IN(37);
  const float* wgt0 = (const float*)IN(38);
  const float* sn0 = (const float*)IN(39);
  const float* mw0 = (const float*)IN(40);
  const int* idx1 = (const int*)IN(43);
  const float* wgt1 = (const float*)IN(44);
  const float* sn1 = (const float*)IN(45);
  const float* mw1 = (const float*)IN(46);
  const int* low_idx = (const int*)IN(49);
  const float* w_full = (const float*)IN(50);
  const int Ml = (49 < n_in) ? in_sizes[49] : 1;

  // ------------------------- workspace carve ------------------------------
  char* ws = (char*)d_ws;
  size_t off = 0;
  auto alloc = [&](size_t bytes) -> void* {
    void* p = ws + off;
    off += (bytes + 255) & ~(size_t)255;
    return p;
  };
  float2* yc = (float2*)alloc((size_t)B * M * 8);
  float2* ylow = (float2*)alloc((size_t)B * M * 8);
  float2* ytmp = (float2*)alloc((size_t)B * M * 8);
  float2* yfull = (float2*)alloc((size_t)B * M * 8);
  float2* yres = (float2*)alloc((size_t)B * M * 8);
  float2* grid0 = (float2*)alloc((size_t)B * KK0 * 8);
  float2* grid0b = (float2*)alloc((size_t)B * KK0 * 8);
  float2* grid1 = (float2*)alloc((size_t)B * KK1 * 8);
  float2* grid1b = (float2*)alloc((size_t)B * KK1 * 8);
  float* img_init = (float*)alloc((size_t)P0 * 4);
  float* h1 = (float*)alloc((size_t)P1 * 4 * 4);
  float* h0 = (float*)alloc((size_t)P0 * 4 * 4);
  float* act_a = (float*)alloc((size_t)P0 * 16 * 4);
  float* act_b = (float*)alloc((size_t)P0 * 16 * 4);
  float* x1 = (float*)alloc((size_t)P1 * 4);
  float* x0 = (float*)alloc((size_t)P0 * 4);
  _Float16* w1a = (_Float16*)alloc(64 * 32 * 2);
  _Float16* w1b = (_Float16*)alloc(288 * 32 * 2);
  _Float16* w0a = (_Float16*)alloc(64 * 16 * 2);
  _Float16* w0b = (_Float16*)alloc(160 * 16 * 2);

  // ----------------------------- helpers ----------------------------------
  auto fft2_0 = [&](float2* g, float sgn) {  // 512x512, B batches; inverse unscaled
    k_fft_line<512, 9><<<dim3(B * 512), dim3(256), 0, stream>>>(g, 512, KK0, 512, 1, sgn, 1.f);
    k_fft_line<512, 9><<<dim3(B * 512), dim3(256), 0, stream>>>(g, 512, KK0, 1, 512, sgn, 1.f);
  };
  auto fft2_1 = [&](float2* g, float sgn) {  // 256x256
    k_fft_line<256, 8><<<dim3(B * 256), dim3(128), 0, stream>>>(g, 256, KK1, 256, 1, sgn, 1.f);
    k_fft_line<256, 8><<<dim3(B * 256), dim3(128), 0, stream>>>(g, 256, KK1, 1, 256, sgn, 1.f);
  };
  auto adj0 = [&](const float2* yy, const float* mw, float* outp, int os, long oo) {
    k_fill_zero<<<gb(B * KK0 * 2), 256, 0, stream>>>((float*)grid0b, B * KK0 * 2);
    k_adj_scatter<<<gb((long)B * M), 256, 0, stream>>>(yy, wgt0, idx0, mw, grid0b, M, B, KK0);
    fft2_0(grid0b, +1.f);
    k_crop_sn_real<<<gb(P0), 256, 0, stream>>>(grid0b, sn0, outp, B, 256, 256, 512, KK0, os, oo);
  };
  auto adj1 = [&](const float2* yy, const float* mw, float* outp, int os, long oo) {
    k_fill_zero<<<gb(B * KK1 * 2), 256, 0, stream>>>((float*)grid1b, B * KK1 * 2);
    k_adj_scatter<<<gb((long)B * Ml), 256, 0, stream>>>(yy, wgt1, idx1, mw, grid1b, Ml, B, KK1);
    fft2_1(grid1b, +1.f);
    k_crop_sn_real<<<gb(P1), 256, 0, stream>>>(grid1b, sn1, outp, B, 128, 128, 256, KK1, os, oo);
  };
  auto dirXk1 = [&](const float* ximg, int xs, long xo) {  // -> grid1 holds Xk
    k_fill_zero<<<gb(B * KK1 * 2), 256, 0, stream>>>((float*)grid1, B * KK1 * 2);
    k_pad_sn<<<gb(P1), 256, 0, stream>>>(ximg, sn1, grid1, B, 128, 128, 256, KK1, xs, xo);
    fft2_1(grid1, -1.f);
  };
  auto dirXk0 = [&](const float* ximg, int xs, long xo) {  // -> grid0 holds Xk
    k_fill_zero<<<gb(B * KK0 * 2), 256, 0, stream>>>((float*)grid0, B * KK0 * 2);
    k_pad_sn<<<gb(P0), 256, 0, stream>>>(ximg, sn0, grid0, B, 256, 256, 512, KK0, xs, xo);
    fft2_0(grid0, -1.f);
  };

  // ------------------------------ pipeline --------------------------------
  k_pack_y<<<gb((long)B * M), 256, 0, stream>>>(y_ri, yc, (long)B * M);

  // init = Re(adj(plan0, y * w_full))
  adj0(yc, w_full, img_init, 1, 0);

  // low = y[:, low_idx]
  k_gather_low<<<gb((long)B * Ml), 256, 0, stream>>>(yc, low_idx, ylow, Ml, M, B);

  // x = adj(plan1, low*mw)  == dirty (weighting=True) -> channels 0 and 1 of h1
  adj1(ylow, mw1, h1, 4, 0);
  adj1(ylow, mw1, h1, 4, 1);

  // shared residual for both plan1 grads: res = dir(plan1, x) - low
  dirXk1(h1, 4, 0);
  k_dir_gather<<<gb((long)B * Ml), 256, 0, stream>>>(grid1, wgt1, idx1, ylow, yres, Ml, B, KK1);
  adj1(yres, nullptr, h1, 4, 2);
  adj1(yres, mw1, h1, 4, 3);

  // gbn1 + conv block 1 (4 -> 32 -> 32 -> 1), WMMA f16
  k_bn_act<<<gb(P1 * 4), 256, 0, stream>>>(h1, (const float*)IN(1), (const float*)IN(2),
                                           (const float*)IN(3), (const float*)IN(4), 4, P1 * 4, 0);
  k_cvt_w<<<gb(64 * 32), 256, 0, stream>>>((const float*)IN(9), w1a, 36, 64, 32);
  k_cvt_w<<<gb(288 * 32), 256, 0, stream>>>((const float*)IN(15), w1b, 288, 288, 32);
  k_conv3x3_wmma<4, 32, 64><<<dim3((unsigned)(((long)B * 128 * 8 * 2 + 3) / 4)), 128, 0, stream>>>(
      h1, w1a, (const float*)IN(10), act_a, B, 128, 128);
  k_bn_act<<<gb(P1 * 32), 256, 0, stream>>>(act_a, (const float*)IN(11), (const float*)IN(12),
                                            (const float*)IN(13), (const float*)IN(14), 32,
                                            P1 * 32, 1);
  k_conv3x3_wmma<32, 32, 288><<<dim3((unsigned)(((long)B * 128 * 8 * 2 + 3) / 4)), 128, 0,
                                stream>>>(act_a, w1b, (const float*)IN(16), act_b, B, 128, 128);
  k_bn_act<<<gb(P1 * 32), 256, 0, stream>>>(act_b, (const float*)IN(17), (const float*)IN(18),
                                            (const float*)IN(19), (const float*)IN(20), 32,
                                            P1 * 32, 1);
  k_conv1x1_sig<<<gb(P1), 256, 0, stream>>>(act_b, (const float*)IN(21), (const float*)IN(22), x1,
                                            32, P1);

  // m = dir(plan1, x1);  full = y with low entries replaced by m
  dirXk1(x1, 1, 0);
  k_dir_gather<<<gb((long)B * Ml), 256, 0, stream>>>(grid1, wgt1, idx1, nullptr, ytmp, Ml, B, KK1);
  k_copy_c<<<gb((long)B * M), 256, 0, stream>>>(yc, yfull, (long)B * M);
  k_scatter_full<<<gb((long)B * Ml), 256, 0, stream>>>(yfull, ytmp, low_idx, Ml, M, B);

  // x = adj(plan0, full) -> h0 ch0 ; dirty0 = adj(plan0, y, weighted) -> ch1
  adj0(yfull, nullptr, h0, 4, 0);
  adj0(yc, mw0, h0, 4, 1);

  // shared residual for plan0 grads: res = dir(plan0, x) - y
  dirXk0(h0, 4, 0);
  k_dir_gather<<<gb((long)B * M), 256, 0, stream>>>(grid0, wgt0, idx0, yc, yres, M, B, KK0);
  adj0(yres, nullptr, h0, 4, 2);
  adj0(yres, mw0, h0, 4, 3);

  // gbn0 + conv block 0 (4 -> 16 -> 16 -> 1), WMMA f16
  k_bn_act<<<gb(P0 * 4), 256, 0, stream>>>(h0, (const float*)IN(5), (const float*)IN(6),
                                           (const float*)IN(7), (const float*)IN(8), 4, P0 * 4, 0);
  k_cvt_w<<<gb(64 * 16), 256, 0, stream>>>((const float*)IN(23), w0a, 36, 64, 16);
  k_cvt_w<<<gb(160 * 16), 256, 0, stream>>>((const float*)IN(29), w0b, 144, 160, 16);
  k_conv3x3_wmma<4, 16, 64><<<dim3((unsigned)(((long)B * 256 * 16 + 3) / 4)), 128, 0, stream>>>(
      h0, w0a, (const float*)IN(24), act_a, B, 256, 256);
  k_bn_act<<<gb(P0 * 16), 256, 0, stream>>>(act_a, (const float*)IN(25), (const float*)IN(26),
                                            (const float*)IN(27), (const float*)IN(28), 16,
                                            P0 * 16, 1);
  k_conv3x3_wmma<16, 16, 160><<<dim3((unsigned)(((long)B * 256 * 16 + 3) / 4)), 128, 0, stream>>>(
      act_a, w0b, (const float*)IN(30), act_b, B, 256, 256);
  k_bn_act<<<gb(P0 * 16), 256, 0, stream>>>(act_b, (const float*)IN(31), (const float*)IN(32),
                                            (const float*)IN(33), (const float*)IN(34), 16,
                                            P0 * 16, 1);
  k_conv1x1_sig<<<gb(P0), 256, 0, stream>>>(act_b, (const float*)IN(35), (const float*)IN(36), x0,
                                            16, P0);

  // out = init + x
  k_add<<<gb(out_size), 256, 0, stream>>>(img_init, x0, (float*)d_out, out_size);
}
